// BidirectionalAttention_1632087572541
// MI455X (gfx1250) — matrix operand order repaired and checked by reading, not compiled, not run on hardware
//
#include <hip/hip_runtime.h>
#include <hip/hip_bf16.h>

// ---------------- types ----------------
typedef __attribute__((ext_vector_type(16))) __bf16 v16bf;
typedef __attribute__((ext_vector_type(8)))  float  v8f;
typedef __attribute__((ext_vector_type(4)))  unsigned u32x4;
typedef __attribute__((ext_vector_type(8)))  int      i32x8;
typedef __attribute__((ext_vector_type(4)))  int      i32x4;

union Frag { v16bf v; uint4 u[2]; unsigned w[8]; };

#define BB 8
#define NN 256
#define CC 768
#define HH 12
#define DD 64
#define BH (BB*HH)       // 96
#define BN (BB*NN)       // 2048

#if __has_builtin(__builtin_amdgcn_tensor_load_to_lds) && __has_builtin(__builtin_amdgcn_s_wait_tensorcnt)
#define TDM_OK 1
#else
#define TDM_OK 0
#endif

__device__ __forceinline__ v8f wmma_bf16(v16bf a, v16bf b, v8f c) {
  return __builtin_amdgcn_wmma_f32_16x16x32_bf16(false, a, false, b, (short)0, c, false, false);
}

// A fragment: 16x32 (MxK) tile, row-major, leading dim lda (elements).
__device__ __forceinline__ v16bf load_a_frag(const __bf16* a, int lda) {
  int lane = threadIdx.x & 31;
  int row  = lane & 15;
  int kh   = (lane >> 4) << 3;
  Frag f;
  f.u[0] = *(const uint4*)(a + (long)row * lda + kh);
  f.u[1] = *(const uint4*)(a + (long)row * lda + kh + 16);
  return f.v;
}

// B fragment from row-major W[ncols][K] (out = A @ W^T), global memory.
__device__ __forceinline__ v16bf load_b_frag(const __bf16* w, int ldw) {
  int lane = threadIdx.x & 31;
  int col  = lane & 15;
  int kb   = (lane >> 4) << 3;
  Frag f;
  f.u[0] = *(const uint4*)(w + (long)col * ldw + kb);
  f.u[1] = *(const uint4*)(w + (long)col * ldw + kb + 16);
  return f.v;
}

// B fragment from an LDS-staged 64x32 tile (row-major, 32-elem rows). t = n-subtile.
__device__ __forceinline__ v16bf load_b_frag_lds(const __bf16* lds, int t) {
  int lane = threadIdx.x & 31;
  int col  = (t << 4) + (lane & 15);
  int kb   = (lane >> 4) << 3;
  const __bf16* p = lds + col * 32 + kb;
  Frag f;
  f.u[0] = *(const uint4*)(p);
  f.u[1] = *(const uint4*)(p + 16);
  return f.v;
}

__device__ __forceinline__ float gelu_exact(float x) {
  return 0.5f * x * (1.0f + erff(x * 0.70710678118654752f));
}

#if TDM_OK
// Issue a TDM 2D tile load: tileX elems (bf16) x tileY rows, row stride = strideElems.
// D# bitfields per CDNA5 ISA ch.8 (group0: count/lds/global/type; group1: dims/strides).
// Toolchain uses the 6-arg builtin: (g0, g1, g2, g3, extra_g, cpol).
__device__ __forceinline__ void tdm_load_2d(unsigned ldsByteAddr, const __bf16* gptr,
                                            unsigned tileX, unsigned tileY, long strideElems) {
  unsigned long long ga = (unsigned long long)(size_t)gptr;
  u32x4 g0;
  g0[0] = 1u;                                            // count=1, user desc
  g0[1] = ldsByteAddr;                                   // lds_addr [63:32]
  g0[2] = (unsigned)(ga & 0xFFFFFFFFu);                  // global_addr lo
  g0[3] = (unsigned)((ga >> 32) & 0x1FFFFFFu) | (2u << 30); // global_addr hi | type=2
  unsigned long long sb = (unsigned long long)strideElems;
  i32x8 g1;
  g1[0] = 0x00010000;                                    // wg_mask=0, data_size=1 (2B)
  g1[1] = (int)((tileX & 0xFFFFu) << 16);                // tensor_dim0 lo16 -> [31:16]
  g1[2] = (int)((tileX >> 16) | ((tileY & 0xFFFFu) << 16)); // dim0 hi | dim1 lo
  g1[3] = (int)((tileY >> 16) | ((tileX & 0xFFFFu) << 16)); // dim1 hi | tile_dim0
  g1[4] = (int)(tileY & 0xFFFFu);                        // tile_dim1 (tile_dim2=0)
  g1[5] = (int)(sb & 0xFFFFFFFFu);                       // tensor_dim0_stride lo
  g1[6] = (int)((sb >> 32) & 0xFFFFu);                   // stride hi (dim1_stride=0)
  g1[7] = 0;
  i32x4 z4 = {0, 0, 0, 0};
  i32x8 z8 = {0, 0, 0, 0, 0, 0, 0, 0};
  __builtin_amdgcn_tensor_load_to_lds(g0, g1, z4, z4, z8, 0);
}
#endif

// ---------------- TDM-staged bf16 GEMM (batch=1): out = act(A @ W^T + bias) ----------------
// grid: (N/64, M/128)  block: 256 (8 waves). B tile (64 x 32) staged in LDS, double-buffered.
__global__ void gemm_tdm_k(const __bf16* __restrict__ A, const __bf16* __restrict__ W,
                           const float* __restrict__ bias,
                           float* __restrict__ outF, __bf16* __restrict__ outB,
                           int M, int N, int K, int lda, int ldw, int ldo, int act) {
  __shared__ __attribute__((aligned(128))) __bf16 bufS[2][64 * 32];
  int wave  = threadIdx.x >> 5;
  int mBase = blockIdx.y * 128 + wave * 16;
  int nBase = blockIdx.x * 64;
  int nc = K >> 5;

  auto stage = [&](int c, int buf) {
    const __bf16* src = W + (long)nBase * ldw + c * 32;
#if TDM_OK
    if (wave == 0)
      tdm_load_2d((unsigned)(size_t)(void*)&bufS[buf][0], src, 32u, 64u, (long)ldw);
#else
    int t = threadIdx.x;          // 2048 elems / 256 thr = one uint4 each
    int row = t >> 2, seg = t & 3;
    *(uint4*)(&bufS[buf][row * 32 + seg * 8]) = *(const uint4*)(src + (long)row * ldw + seg * 8);
#endif
  };

  stage(0, 0);
  v8f acc[4] = {};
  for (int c = 0; c < nc; ++c) {
    if (c + 1 < nc) {
      stage(c + 1, (c + 1) & 1);
#if TDM_OK
      if (wave == 0) __builtin_amdgcn_s_wait_tensorcnt(1);
#endif
    } else {
#if TDM_OK
      if (wave == 0) __builtin_amdgcn_s_wait_tensorcnt(0);
#endif
    }
    __syncthreads();
    v16bf af = load_a_frag(A + (long)mBase * lda + c * 32, lda);
    const __bf16* bt = &bufS[c & 1][0];
#pragma unroll
    for (int t = 0; t < 4; ++t)
      acc[t] = wmma_bf16(af, load_b_frag_lds(bt, t), acc[t]);
    __syncthreads();
  }

  int lane = threadIdx.x & 31;
  int colL = lane & 15;
  int rowB = (lane >> 4) << 3;
#pragma unroll
  for (int t = 0; t < 4; ++t) {
    int n = nBase + t * 16 + colL;
    float bv = bias ? bias[n] : 0.0f;
#pragma unroll
    for (int i = 0; i < 8; ++i) {
      int m = mBase + rowB + i;
      float v = acc[t][i] + bv;
      if (act == 1) v = gelu_exact(v);
      long o = (long)m * ldo + n;
      if (outF) outF[o] = v;
      if (outB) outB[o] = (__bf16)v;
    }
  }
}

// ---------------- direct batched bf16 GEMM: out = act(scale * A @ W^T + bias) ----------------
// grid: (N/64, M/128, numBatch)  block: 256
__global__ void gemm_bf16_k(const __bf16* __restrict__ A, const __bf16* __restrict__ W,
                            const float* __restrict__ bias,
                            float* __restrict__ outF, __bf16* __restrict__ outB,
                            int M, int N, int K, int lda, int ldw, int ldo,
                            long aSB, long aSH, long wSB, long wSH, long oSB, long oSH,
                            int Hdiv, float scale, int act) {
  int z = blockIdx.z;
  int b = z / Hdiv, h = z % Hdiv;
  A += (long)b * aSB + (long)h * aSH;
  W += (long)b * wSB + (long)h * wSH;
  long oOff = (long)b * oSB + (long)h * oSH;

  int wave  = threadIdx.x >> 5;
  int mBase = blockIdx.y * 128 + wave * 16;
  int nBase = blockIdx.x * 64;
  if (mBase >= M) return;

  v8f acc[4] = {};
  for (int k = 0; k < K; k += 32) {
    v16bf af = load_a_frag(A + (long)mBase * lda + k, lda);
#pragma unroll
    for (int t = 0; t < 4; ++t) {
      v16bf bf = load_b_frag(W + (long)(nBase + t * 16) * ldw + k, ldw);
      acc[t] = wmma_bf16(af, bf, acc[t]);
    }
  }
  int lane = threadIdx.x & 31;
  int colL = lane & 15;
  int rowB = (lane >> 4) << 3;
#pragma unroll
  for (int t = 0; t < 4; ++t) {
    int n = nBase + t * 16 + colL;
    float bv = bias ? bias[n] : 0.0f;
#pragma unroll
    for (int i = 0; i < 8; ++i) {
      int m = mBase + rowB + i;
      float v = acc[t][i] * scale + bv;
      if (act == 1) v = gelu_exact(v);
      long o = oOff + (long)m * ldo + n;
      if (outF) outF[o] = v;
      if (outB) outB[o] = (__bf16)v;
    }
  }
}

// ---------------- fused isa path: isa = (li x gi outer) @ W2^T + gs @ w_bg^T ----------------
// grid: (16 token-tiles, 96 bh)  block: 128 (4 waves; wave -> 16 e-columns)
__global__ void isa_fused_k(const __bf16* __restrict__ gi, const __bf16* __restrict__ li,
                            const __bf16* __restrict__ w2, const __bf16* __restrict__ gs,
                            const __bf16* __restrict__ wbg, float* __restrict__ isa) {
  int z = blockIdx.y;
  int b = z / HH, h = z % HH;
  int tokBase = blockIdx.x * 16;
  const __bf16* giP = gi + ((long)(b * NN + tokBase)) * CC + h * DD;
  const __bf16* liP = li + ((long)(b * NN + tokBase)) * CC + h * DD;
  const __bf16* gsP = gs + ((long)z * NN + tokBase) * DD;

  __shared__ unsigned short liS[16 * 64];
  for (int i = threadIdx.x; i < 16 * 64; i += blockDim.x) {
    int r = i >> 6, c = i & 63;
    liS[i] = __builtin_bit_cast(unsigned short, liP[(long)r * CC + c]);
  }
  __syncthreads();

  int wave  = threadIdx.x >> 5;
  int lane  = threadIdx.x & 31;
  int row   = lane & 15;
  int eBase = wave * 16;

  // gi A-fragments for k' in [0,32) and [32,64): only two variants over whole K loop
  Frag g0, g1;
  g0.v = load_a_frag(giP, CC);
  g1.v = load_a_frag(giP + 32, CC);

  v8f acc = {};
  for (int kt = 0; kt < 128; ++kt) {      // K = 4096, kk = d*64 + k'
    int d = kt >> 1;
    unsigned short us = liS[row * 64 + d];
    unsigned lpack = (unsigned)us | ((unsigned)us << 16);  // packed bf16 pair
    const Frag& gsel = (kt & 1) ? g1 : g0;
    Frag a;
#pragma unroll
    for (int i = 0; i < 8; ++i)
      asm("v_pk_mul_bf16 %0, %1, %2" : "=v"(a.w[i]) : "v"(gsel.w[i]), "v"(lpack));
    v16bf bf = load_b_frag(w2 + (long)eBase * 4096 + kt * 32, 4096);
    acc = wmma_bf16(a.v, bf, acc);
  }
  // fused bias path: + gs(16x64) @ w_bg^T
#pragma unroll
  for (int k = 0; k < 64; k += 32) {
    v16bf a  = load_a_frag(gsP + k, DD);
    v16bf bf = load_b_frag(wbg + (long)eBase * DD + k, DD);
    acc = wmma_bf16(a, bf, acc);
  }

  int e    = eBase + (lane & 15);
  int rowB = (lane >> 4) << 3;
#pragma unroll
  for (int i = 0; i < 8; ++i) {
    int m = tokBase + rowB + i;
    isa[((long)(b * NN) + m) * CC + h * DD + e] = acc[i];
  }
}

// ---------------- softmax over last dim (rows) ----------------
__global__ void softmax_row_k(const float* __restrict__ lg, __bf16* __restrict__ out) {
  long row = blockIdx.x;
  const float* p = lg + row * NN;
  int t = threadIdx.x;
  float x = p[t];
  __shared__ float red[NN];
  red[t] = x; __syncthreads();
  for (int s = 128; s > 0; s >>= 1) { if (t < s) red[t] = fmaxf(red[t], red[t + s]); __syncthreads(); }
  float mx = red[0]; __syncthreads();
  float e = __expf(x - mx);
  red[t] = e; __syncthreads();
  for (int s = 128; s > 0; s >>= 1) { if (t < s) red[t] += red[t + s]; __syncthreads(); }
  out[row * NN + t] = (__bf16)(e / red[0]);
}

// ---------------- softmax over axis=-2 (columns) ----------------
__global__ void softmax_col_k(const float* __restrict__ lg, __bf16* __restrict__ out) {
  int bh = blockIdx.x >> 8, m = blockIdx.x & 255;
  const float* p = lg + (long)bh * NN * NN + m;
  int t = threadIdx.x;
  float x = p[(long)t * NN];
  __shared__ float red[NN];
  red[t] = x; __syncthreads();
  for (int s = 128; s > 0; s >>= 1) { if (t < s) red[t] = fmaxf(red[t], red[t + s]); __syncthreads(); }
  float mx = red[0]; __syncthreads();
  float e = __expf(x - mx);
  red[t] = e; __syncthreads();
  for (int s = 128; s > 0; s >>= 1) { if (t < s) red[t] += red[t + s]; __syncthreads(); }
  out[(long)bh * NN * NN + (long)t * NN + m] = (__bf16)(e / red[0]);
}

// ---------------- per-head transpose [N,D] -> [D,N] (bf16) ----------------
__global__ void transpose_head_k(const __bf16* __restrict__ in, __bf16* __restrict__ out,
                                 long inSB, long inSH, int Hdiv, int rs) {
  int z = blockIdx.x;
  int b = z / Hdiv, h = z % Hdiv;
  const __bf16* p = in + (long)b * inSB + (long)h * inSH;
  __bf16* o = out + (long)z * (DD * NN);
  int t = threadIdx.x;
  for (int d = 0; d < DD; ++d) o[d * NN + t] = p[(long)t * rs + d];
}

// ---------------- LayerNorm x2 + sigmoid gate -> bf16 ----------------
__global__ void ln_gate_k(const float* __restrict__ sa, const float* __restrict__ isa,
                          const float* __restrict__ lam, __bf16* __restrict__ xc) {
  long row = blockIdx.x;
  const float* ps = sa + row * CC;
  const float* pi = isa + row * CC;
  int t = threadIdx.x;
  float sv[3], iv[3], s1 = 0, s2 = 0, i1 = 0, i2 = 0;
#pragma unroll
  for (int j = 0; j < 3; ++j) {
    float a = ps[t + j * 256], b2 = pi[t + j * 256];
    sv[j] = a; iv[j] = b2;
    s1 += a; s2 += a * a; i1 += b2; i2 += b2 * b2;
  }
  __shared__ float r1[256], r2[256], r3[256], r4[256];
  r1[t] = s1; r2[t] = s2; r3[t] = i1; r4[t] = i2; __syncthreads();
  for (int s = 128; s > 0; s >>= 1) {
    if (t < s) { r1[t] += r1[t + s]; r2[t] += r2[t + s]; r3[t] += r3[t + s]; r4[t] += r4[t + s]; }
    __syncthreads();
  }
  float ms = r1[0] / CC, vs = r2[0] / CC - ms * ms;
  float mi = r3[0] / CC, vi = r4[0] / CC - mi * mi;
  float rs = rsqrtf(vs + 1e-5f), ri = rsqrtf(vi + 1e-5f);
  float g = 1.0f / (1.0f + __expf(-lam[0]));
#pragma unroll
  for (int j = 0; j < 3; ++j) {
    float o = (sv[j] - ms) * rs * g + (iv[j] - mi) * ri * (1.0f - g);
    xc[row * CC + t + j * 256] = (__bf16)o;
  }
}

// ---------------- fp32 -> bf16 ----------------
__global__ void cvt_bf16_k(const float* __restrict__ in, __bf16* __restrict__ out, int n) {
  int i = blockIdx.x * blockDim.x + threadIdx.x;
  if (i < n) out[i] = (__bf16)in[i];
}

// ---------------- permute w_g[d*64+e, k] -> W2[e, d*64+k] (bf16) ----------------
__global__ void permute_wg_k(const float* __restrict__ wg, __bf16* __restrict__ w2) {
  int o = blockIdx.x * blockDim.x + threadIdx.x;
  if (o >= 64 * 64 * 64) return;
  int e = o >> 12, d = (o >> 6) & 63, k = o & 63;
  w2[o] = (__bf16)wg[((d << 6) + e) * 64 + k];
}

// ---------------- host launch ----------------
extern "C" void kernel_launch(void* const* d_in, const int* in_sizes, int n_in,
                              void* d_out, int out_size, void* d_ws, size_t ws_size,
                              hipStream_t stream) {
  const float* x        = (const float*)d_in[0];
  const float* w_qkv    = (const float*)d_in[1];
  const float* b_qkv    = (const float*)d_in[2];
  const float* w_g      = (const float*)d_in[3];
  const float* w_bg     = (const float*)d_in[4];
  const float* w_local  = (const float*)d_in[5];
  const float* b_local  = (const float*)d_in[6];
  const float* w_global = (const float*)d_in[7];
  const float* b_global = (const float*)d_in[8];
  const float* lam      = (const float*)d_in[9];
  const float* w_proj   = (const float*)d_in[10];
  const float* b_proj   = (const float*)d_in[11];
  float* out = (float*)d_out;

  char* ws = (char*)d_ws;
  size_t off = 0;
  auto alloc = [&](size_t bytes) { void* p = ws + off; off = (off + bytes + 255) & ~(size_t)255; return p; };

  __bf16* xb       = (__bf16*)alloc((size_t)BN * CC * 2);
  __bf16* wqkv_b   = (__bf16*)alloc((size_t)3 * CC * CC * 2);
  __bf16* wloc_b   = (__bf16*)alloc((size_t)CC * CC * 2);
  __bf16* wglb_b   = (__bf16*)alloc((size_t)CC * CC * 2);
  __bf16* wprj_b   = (__bf16*)alloc((size_t)CC * CC * 2);
  __bf16* wbg_b    = (__bf16*)alloc((size_t)DD * DD * 2);
  __bf16* w2_b     = (__bf16*)alloc((size_t)64 * 4096 * 2);
  __bf16* qkv_b    = (__bf16*)alloc((size_t)BN * 3 * CC * 2);
  float*  logits_f = (float*)alloc((size_t)BH * NN * NN * 4);
  __bf16* attn_b   = (__bf16*)alloc((size_t)BH * NN * NN * 2);
  __bf16* rattn_b  = (__bf16*)alloc((size_t)BH * NN * NN * 2);
  __bf16* gi_b     = (__bf16*)alloc((size_t)BN * CC * 2);
  __bf16* li_b     = (__bf16*)alloc((size_t)BN * CC * 2);
  __bf16* vt_b     = (__bf16*)alloc((size_t)BH * DD * NN * 2);
  __bf16* giT_b    = (__bf16*)alloc((size_t)BH * DD * NN * 2);
  __bf16* gs_b     = (__bf16*)alloc((size_t)BH * NN * DD * 2);
  float*  sa_f     = (float*)alloc((size_t)BN * CC * 4);
  float*  isa_f    = (float*)alloc((size_t)BN * CC * 4);
  __bf16* xc_b     = (__bf16*)alloc((size_t)BN * CC * 2);
  (void)ws_size; (void)n_in; (void)in_sizes; (void)out_size;

  auto cvt = [&](const float* src, __bf16* dst, int n) {
    cvt_bf16_k<<<(n + 255) / 256, 256, 0, stream>>>(src, dst, n);
  };
  cvt(x, xb, BN * CC);
  cvt(w_qkv, wqkv_b, 3 * CC * CC);
  cvt(w_local, wloc_b, CC * CC);
  cvt(w_global, wglb_b, CC * CC);
  cvt(w_proj, wprj_b, CC * CC);
  cvt(w_bg, wbg_b, DD * DD);
  permute_wg_k<<<(64 * 64 * 64 + 255) / 256, 256, 0, stream>>>(w_g, w2_b);

  // qkv = x @ w_qkv^T + b_qkv -> bf16 [2048, 2304]   (TDM-staged B)
  gemm_tdm_k<<<dim3(36, 16), 256, 0, stream>>>(
      xb, wqkv_b, b_qkv, nullptr, qkv_b, BN, 3 * CC, CC, CC, CC, 3 * CC, 0);
  // gi = gelu(x @ w_global^T + b_global) -> bf16
  gemm_tdm_k<<<dim3(12, 16), 256, 0, stream>>>(
      xb, wglb_b, b_global, nullptr, gi_b, BN, CC, CC, CC, CC, CC, 1);
  // li = gelu(x @ w_local^T + b_local) -> bf16
  gemm_tdm_k<<<dim3(12, 16), 256, 0, stream>>>(
      xb, wloc_b, b_local, nullptr, li_b, BN, CC, CC, CC, CC, CC, 1);

  // logits[b,h] = scale * Q K^T  (q at col 0, k at col CC; batched over 96)
  gemm_bf16_k<<<dim3(4, 2, BH), 256, 0, stream>>>(
      qkv_b, qkv_b + CC, nullptr, logits_f, nullptr, NN, NN, DD, 3 * CC, 3 * CC, NN,
      (long)NN * 3 * CC, DD, (long)NN * 3 * CC, DD, (long)HH * NN * NN, (long)NN * NN,
      HH, 0.125f, 0);

  softmax_row_k<<<BH * NN, 256, 0, stream>>>(logits_f, attn_b);
  softmax_col_k<<<BH * NN, 256, 0, stream>>>(logits_f, rattn_b);

  // V^T and gi^T per head
  transpose_head_k<<<BH, 256, 0, stream>>>(qkv_b + 2 * CC, vt_b,
                                           (long)NN * 3 * CC, DD, HH, 3 * CC);
  transpose_head_k<<<BH, 256, 0, stream>>>(gi_b, giT_b,
                                           (long)NN * CC, DD, HH, CC);

  // sa[b,h] = attn @ V -> fp32 [B,N,C] layout
  gemm_bf16_k<<<dim3(1, 2, BH), 256, 0, stream>>>(
      attn_b, vt_b, nullptr, sa_f, nullptr, NN, DD, NN, NN, NN, CC,
      (long)HH * NN * NN, (long)NN * NN, (long)HH * DD * NN, (long)DD * NN,
      (long)NN * CC, DD, HH, 1.0f, 0);
  // gs[b,h] = rattn @ gi -> bf16 [B,H,N,D]
  gemm_bf16_k<<<dim3(1, 2, BH), 256, 0, stream>>>(
      rattn_b, giT_b, nullptr, nullptr, gs_b, NN, DD, NN, NN, NN, DD,
      (long)HH * NN * NN, (long)NN * NN, (long)HH * DD * NN, (long)DD * NN,
      (long)HH * NN * DD, (long)NN * DD, HH, 1.0f, 0);

  // isa = P @ W2^T + gs @ w_bg^T (fused, on-the-fly outer-product A via v_pk_mul_bf16)
  isa_fused_k<<<dim3(16, BH), 128, 0, stream>>>(gi_b, li_b, w2_b, gs_b, wbg_b, isa_f);

  // LayerNorm both paths + gate -> bf16
  ln_gate_k<<<BN, 256, 0, stream>>>(sa_f, isa_f, lam, xc_b);

  // out = xc @ w_proj^T + b_proj (fp32 out, TDM-staged B)
  gemm_tdm_k<<<dim3(12, 16), 256, 0, stream>>>(
      xc_b, wprj_b, b_proj, out, nullptr, BN, CC, CC, CC, CC, CC, 0);
}
